// CausalSelfAttention_61512521613658
// MI455X (gfx1250) — compile-verified, matrix operand order checked by reading, not run on hardware
//
#include <hip/hip_runtime.h>
#include <hip/hip_bf16.h>
#include <stdint.h>

// ---------------------------------------------------------------------------
// Causal self-attention fwd on gfx1250 (MI455X), bf16 WMMA, flash-style,
// async global->LDS double-buffered GEMM staging.
// B=4 T=2048 C=1024 H=16 hd=64
// ---------------------------------------------------------------------------

typedef __bf16 bf16x16 __attribute__((ext_vector_type(16)));
typedef float  f32x8   __attribute__((ext_vector_type(8)));
typedef int    v4i     __attribute__((vector_size(16)));   // matches builtin proto

#define B_  4
#define T_  2048
#define C_  1024
#define H_  16
#define HD_ 64
#define M_  (B_ * T_)   // 8192 rows
#define N1_ (3 * C_)    // 3072 qkv cols

#define LDT  48         // padded LDS stride (elements) -> 96B rows, 16B aligned
#define LBUF (128 * LDT)

#define AS1 __attribute__((address_space(1)))
#define AS3 __attribute__((address_space(3)))

__device__ __forceinline__ unsigned short f2bf(float f) {
  unsigned int u = __float_as_uint(f);
  u += 0x7FFFu + ((u >> 16) & 1u);   // round-to-nearest-even
  return (unsigned short)(u >> 16);
}

union FragU { bf16x16 f; uint4 u[2]; };

// Load one 16x32 bf16 fragment (A or B operand) from a K-contiguous matrix.
// lane L: r = L&15, half = L>>4; VGPR v: K = (v>>2)*16 + half*8 + (v&3)*2.
__device__ __forceinline__ bf16x16 frag_ld(const unsigned short* p, int stride,
                                           int row, int kbase) {
  const int lane = threadIdx.x & 31;
  const int r = lane & 15, h = lane >> 4;
  const unsigned short* base = p + (size_t)(row + r) * stride + kbase + h * 8;
  FragU fr;
  fr.u[0] = *(const uint4*)(base);        // K = half*8 .. half*8+7
  fr.u[1] = *(const uint4*)(base + 16);   // K = 16+half*8 .. 16+half*8+7
  return fr.f;
}

__device__ __forceinline__ f32x8 wmma_bf(bf16x16 a, bf16x16 b, f32x8 c) {
  return __builtin_amdgcn_wmma_f32_16x16x32_bf16(false, a, false, b,
                                                 (short)0, c, false, false);
}

// 16B async copy global->LDS (CDNA5 GLOBAL_LOAD_ASYNC_TO_LDS_B128, ASYNCcnt).
// Address-space casts go through integers: generic LDS ptrs carry the LDS
// offset in the low 32 bits (ISA aperture rules); AS1 is value-identical.
// AS3 pointers are 32-bit on amdgcn, so the uint round-trip is pointer-sized.
__device__ __forceinline__ void async_cp128(const unsigned short* g, unsigned short* l) {
#if __has_builtin(__builtin_amdgcn_global_load_async_to_lds_b128)
  __builtin_amdgcn_global_load_async_to_lds_b128(
      (AS1 v4i*)(unsigned long long)(uintptr_t)g,
      (AS3 v4i*)(unsigned int)(uintptr_t)l, 0, 0);
#else
  *(uint4*)l = *(const uint4*)g;
#endif
}

__device__ __forceinline__ void wait_async() {
#if __has_builtin(__builtin_amdgcn_s_wait_asynccnt)
  __builtin_amdgcn_s_wait_asynccnt(0);
#else
  asm volatile("s_wait_asynccnt 0" ::: "memory");
#endif
}

// ---------------------------------------------------------------------------
// Prep kernels: fp32 -> bf16, and transposed bf16 weights (N-major).
// ---------------------------------------------------------------------------
__global__ void k_cvt(const float* __restrict__ in, unsigned short* __restrict__ out, int n) {
  int i = blockIdx.x * blockDim.x + threadIdx.x;
  if (i < n) out[i] = f2bf(in[i]);
}

// in[R][Nc] fp32 -> out[Nc][R] bf16
__global__ void k_transpose(const float* __restrict__ in, unsigned short* __restrict__ out,
                            int R, int Nc) {
  long long i = (long long)blockIdx.x * blockDim.x + threadIdx.x;
  if (i < (long long)R * Nc) {
    int r = (int)(i / Nc), c = (int)(i % Nc);
    out[(long long)c * R + r] = f2bf(in[i]);
  }
}

// ---------------------------------------------------------------------------
// GEMM1: qkv = x @ w_attn + b_attn, scattered into Q[B,H,T,hd], K[B,H,T,hd],
// Vt[B,H,hd,T]. 128x128 block tile, 256 threads = 8 waves (2x4),
// double-buffered async LDS staging, 8 WMMAs per wave per K-step.
// ---------------------------------------------------------------------------
__global__ __launch_bounds__(256) void gemm_qkv(
    const unsigned short* __restrict__ A,   // x bf16 [M_, C_]
    const unsigned short* __restrict__ Bt,  // w_attn^T bf16 [N1_, C_]
    const float* __restrict__ bias,         // [N1_]
    unsigned short* __restrict__ Qo, unsigned short* __restrict__ Ko,
    unsigned short* __restrict__ Vt) {
  __shared__ unsigned short lA[2 * LBUF];
  __shared__ unsigned short lB[2 * LBUF];
  const int tid = threadIdx.x;
  const int lane = tid & 31, wave = tid >> 5;
  const int wm = wave >> 2, wn = wave & 3;           // 2 x 4 waves
  const int mbase = blockIdx.x * 128, nbase = blockIdx.y * 128;
  const int rowh = tid >> 2, quad = tid & 3;         // staging: 4 threads/row

  auto stage = [&](int k0, int buf) {
#pragma unroll
    for (int i = 0; i < 2; ++i) {
      const int row = rowh + i * 64;
      async_cp128(A + (size_t)(mbase + row) * C_ + k0 + quad * 8,
                  lA + buf * LBUF + row * LDT + quad * 8);
      async_cp128(Bt + (size_t)(nbase + row) * C_ + k0 + quad * 8,
                  lB + buf * LBUF + row * LDT + quad * 8);
    }
  };

  const f32x8 fzero = {};
  f32x8 acc[4][2];
#pragma unroll
  for (int i = 0; i < 4; ++i)
#pragma unroll
    for (int j = 0; j < 2; ++j) acc[i][j] = fzero;

  stage(0, 0);
  int cur = 0;
  for (int k0 = 0; k0 < C_; k0 += 32) {
    wait_async();        // my tile loads landed
    __syncthreads();     // everyone's tile loads landed
    if (k0 + 32 < C_) {
      stage(k0 + 32, cur ^ 1);
      if (k0 + 64 < C_) {  // global_prefetch_b8 two tiles ahead
        __builtin_prefetch(A + (size_t)(mbase + rowh) * C_ + k0 + 64 + quad * 8, 0, 1);
        __builtin_prefetch(Bt + (size_t)(nbase + rowh) * C_ + k0 + 64 + quad * 8, 0, 1);
      }
    }
    const unsigned short* sA = lA + cur * LBUF;
    const unsigned short* sB = lB + cur * LBUF;
    bf16x16 af[4], bfm[2];
#pragma unroll
    for (int i = 0; i < 4; ++i) af[i] = frag_ld(sA, LDT, wm * 64 + i * 16, 0);
#pragma unroll
    for (int j = 0; j < 2; ++j) bfm[j] = frag_ld(sB, LDT, wn * 32 + j * 16, 0);
#pragma unroll
    for (int i = 0; i < 4; ++i)
#pragma unroll
      for (int j = 0; j < 2; ++j) acc[i][j] = wmma_bf(af[i], bfm[j], acc[i][j]);
    cur ^= 1;
  }

  // Epilogue: bias add, bf16 convert, scatter into Q/K/Vt layouts.
  const int nlo = lane & 15, half = lane >> 4;
#pragma unroll
  for (int i = 0; i < 4; ++i) {
#pragma unroll
    for (int j = 0; j < 2; ++j) {
      const int n = nbase + wn * 32 + j * 16 + nlo;
      const float bv = bias[n];
      const int which = n >> 10, c = n & 1023, h = c >> 6, d = c & 63;
#pragma unroll
      for (int r = 0; r < 8; ++r) {
        const int m = mbase + wm * 64 + i * 16 + r + half * 8;
        const int b = m >> 11, t = m & 2047;
        const unsigned short val = f2bf(acc[i][j][r] + bv);
        const size_t bh = (size_t)(b * H_ + h);
        if (which == 0)      Qo[(bh * T_ + t) * HD_ + d] = val;
        else if (which == 1) Ko[(bh * T_ + t) * HD_ + d] = val;
        else                 Vt[(bh * HD_ + d) * T_ + t] = val;
      }
    }
  }
}

// ---------------------------------------------------------------------------
// Flash attention: grid (T/64, B*H), 128 threads = 4 waves, one 16-row q tile
// per wave, 32-key steps with online softmax. S via WMMA(Q, K^T), P bounced
// through per-wave LDS (D-layout -> A-layout), O += WMMA(P, V).
// ---------------------------------------------------------------------------
__global__ __launch_bounds__(128) void attn_fwd(
    const unsigned short* __restrict__ Qm, const unsigned short* __restrict__ Km,
    const unsigned short* __restrict__ Vt, unsigned short* __restrict__ Y) {
  __shared__ unsigned short lP[4][16 * 32];   // 1 KB per wave
  const int tid = threadIdx.x, lane = tid & 31, wave = tid >> 5;
  const int bh = blockIdx.y;
  const int qbase = blockIdx.x * 64 + wave * 16;
  const unsigned short* q = Qm + (size_t)bh * T_ * HD_;
  const unsigned short* k = Km + (size_t)bh * T_ * HD_;
  const unsigned short* v = Vt + (size_t)bh * HD_ * T_;
  const int nlo = lane & 15, half = lane >> 4;

  const bf16x16 q0 = frag_ld(q, HD_, qbase, 0);
  const bf16x16 q1 = frag_ld(q, HD_, qbase, 32);

  const f32x8 fzero = {};
  f32x8 o[4];
#pragma unroll
  for (int t = 0; t < 4; ++t) o[t] = fzero;
  float rmax[8], rsum[8];
#pragma unroll
  for (int r = 0; r < 8; ++r) { rmax[r] = -1e30f; rsum[r] = 0.f; }

  unsigned short* lp = lP[wave];
  const int njt = (qbase + 16 + 31) >> 5;   // causal: keys up to qbase+15
  for (int jt = 0; jt < njt; ++jt) {
    const int j0 = jt * 32;
    // S = Q @ K^T for 32 keys: two 16x16 tiles, K-dim 64 = 2 WMMAs each.
    f32x8 s0 = fzero, s1 = fzero;
    s0 = wmma_bf(q0, frag_ld(k, HD_, j0, 0), s0);
    s0 = wmma_bf(q1, frag_ld(k, HD_, j0, 32), s0);
    s1 = wmma_bf(q0, frag_ld(k, HD_, j0 + 16, 0), s1);
    s1 = wmma_bf(q1, frag_ld(k, HD_, j0 + 16, 32), s1);

    // Online softmax per row; reductions within each 16-lane half.
#pragma unroll
    for (int r = 0; r < 8; ++r) {
      const int qrow = qbase + r + half * 8;
      float a0 = (j0 + nlo <= qrow)      ? s0[r] * 0.125f : -1e30f;  // hd^-0.5
      float a1 = (j0 + 16 + nlo <= qrow) ? s1[r] * 0.125f : -1e30f;
      float mt = fmaxf(a0, a1);
      mt = fmaxf(mt, __shfl_xor(mt, 1));
      mt = fmaxf(mt, __shfl_xor(mt, 2));
      mt = fmaxf(mt, __shfl_xor(mt, 4));
      mt = fmaxf(mt, __shfl_xor(mt, 8));
      const float mnew = fmaxf(rmax[r], mt);
      const float corr = __expf(rmax[r] - mnew);
      const float p0 = __expf(a0 - mnew);
      const float p1 = __expf(a1 - mnew);
      float ps = p0 + p1;
      ps += __shfl_xor(ps, 1);
      ps += __shfl_xor(ps, 2);
      ps += __shfl_xor(ps, 4);
      ps += __shfl_xor(ps, 8);
      rsum[r] = rsum[r] * corr + ps;
      rmax[r] = mnew;
      o[0][r] *= corr; o[1][r] *= corr; o[2][r] *= corr; o[3][r] *= corr;
      const int m = r + half * 8;               // D-layout row -> LDS row
      lp[m * 32 + nlo]      = f2bf(p0);
      lp[m * 32 + 16 + nlo] = f2bf(p1);
    }
    asm volatile("s_wait_dscnt 0" ::: "memory");   // P stores visible in-wave

    // O += P @ V : P is 16x32 A-frag, V^T gives K-contiguous B-frags.
    const bf16x16 pf = frag_ld(lp, 32, 0, 0);
#pragma unroll
    for (int t = 0; t < 4; ++t)
      o[t] = wmma_bf(pf, frag_ld(v, T_, t * 16, j0), o[t]);
  }

  // Normalize and store y[B,T,C] bf16.
  const int b = bh >> 4, h = bh & 15;
#pragma unroll
  for (int t = 0; t < 4; ++t) {
#pragma unroll
    for (int r = 0; r < 8; ++r) {
      const int trow = qbase + r + half * 8;
      const float ov = o[t][r] / rsum[r];
      Y[(size_t)(b * T_ + trow) * C_ + h * HD_ + t * 16 + nlo] = f2bf(ov);
    }
  }
}

// ---------------------------------------------------------------------------
// GEMM2: out = y @ w_proj + b_proj, fp32 output. Same double-buffered async
// staging as GEMM1.
// ---------------------------------------------------------------------------
__global__ __launch_bounds__(256) void gemm_proj(
    const unsigned short* __restrict__ A,   // y bf16 [M_, C_]
    const unsigned short* __restrict__ Bt,  // w_proj^T bf16 [C_, C_]
    const float* __restrict__ bias,         // [C_]
    float* __restrict__ Out) {              // [M_, C_] fp32
  __shared__ unsigned short lA[2 * LBUF];
  __shared__ unsigned short lB[2 * LBUF];
  const int tid = threadIdx.x;
  const int lane = tid & 31, wave = tid >> 5;
  const int wm = wave >> 2, wn = wave & 3;
  const int mbase = blockIdx.x * 128, nbase = blockIdx.y * 128;
  const int rowh = tid >> 2, quad = tid & 3;

  auto stage = [&](int k0, int buf) {
#pragma unroll
    for (int i = 0; i < 2; ++i) {
      const int row = rowh + i * 64;
      async_cp128(A + (size_t)(mbase + row) * C_ + k0 + quad * 8,
                  lA + buf * LBUF + row * LDT + quad * 8);
      async_cp128(Bt + (size_t)(nbase + row) * C_ + k0 + quad * 8,
                  lB + buf * LBUF + row * LDT + quad * 8);
    }
  };

  const f32x8 fzero = {};
  f32x8 acc[4][2];
#pragma unroll
  for (int i = 0; i < 4; ++i)
#pragma unroll
    for (int j = 0; j < 2; ++j) acc[i][j] = fzero;

  stage(0, 0);
  int cur = 0;
  for (int k0 = 0; k0 < C_; k0 += 32) {
    wait_async();
    __syncthreads();
    if (k0 + 32 < C_) {
      stage(k0 + 32, cur ^ 1);
      if (k0 + 64 < C_) {
        __builtin_prefetch(A + (size_t)(mbase + rowh) * C_ + k0 + 64 + quad * 8, 0, 1);
        __builtin_prefetch(Bt + (size_t)(nbase + rowh) * C_ + k0 + 64 + quad * 8, 0, 1);
      }
    }
    const unsigned short* sA = lA + cur * LBUF;
    const unsigned short* sB = lB + cur * LBUF;
    bf16x16 af[4], bfm[2];
#pragma unroll
    for (int i = 0; i < 4; ++i) af[i] = frag_ld(sA, LDT, wm * 64 + i * 16, 0);
#pragma unroll
    for (int j = 0; j < 2; ++j) bfm[j] = frag_ld(sB, LDT, wn * 32 + j * 16, 0);
#pragma unroll
    for (int i = 0; i < 4; ++i)
#pragma unroll
      for (int j = 0; j < 2; ++j) acc[i][j] = wmma_bf(af[i], bfm[j], acc[i][j]);
    cur ^= 1;
  }

  const int nlo = lane & 15, half = lane >> 4;
#pragma unroll
  for (int i = 0; i < 4; ++i) {
#pragma unroll
    for (int j = 0; j < 2; ++j) {
      const int n = nbase + wn * 32 + j * 16 + nlo;
      const float bv = bias[n];
#pragma unroll
      for (int r = 0; r < 8; ++r) {
        const int m = mbase + wm * 64 + i * 16 + r + half * 8;
        Out[(size_t)m * C_ + n] = acc[i][j][r] + bv;
      }
    }
  }
}

// ---------------------------------------------------------------------------
// Host launcher. Workspace carve (bytes):
//   xbf 16M | wT 6M | wpT 2M | Q 16M | K 16M | Vt 16M | Y 16M  (~88 MB)
// ---------------------------------------------------------------------------
extern "C" void kernel_launch(void* const* d_in, const int* in_sizes, int n_in,
                              void* d_out, int out_size, void* d_ws, size_t ws_size,
                              hipStream_t stream) {
  const float* x      = (const float*)d_in[0];
  const float* w_attn = (const float*)d_in[1];
  const float* b_attn = (const float*)d_in[2];
  const float* w_proj = (const float*)d_in[3];
  const float* b_proj = (const float*)d_in[4];
  float* out = (float*)d_out;

  char* w = (char*)d_ws;
  unsigned short* xbf = (unsigned short*)w;  w += (size_t)M_ * C_ * 2;
  unsigned short* wT  = (unsigned short*)w;  w += (size_t)N1_ * C_ * 2;
  unsigned short* wpT = (unsigned short*)w;  w += (size_t)C_ * C_ * 2;
  unsigned short* Qb  = (unsigned short*)w;  w += (size_t)M_ * C_ * 2;
  unsigned short* Kb  = (unsigned short*)w;  w += (size_t)M_ * C_ * 2;
  unsigned short* Vtb = (unsigned short*)w;  w += (size_t)M_ * C_ * 2;
  unsigned short* Yb  = (unsigned short*)w;  w += (size_t)M_ * C_ * 2;

  k_cvt<<<(M_ * C_ + 255) / 256, 256, 0, stream>>>(x, xbf, M_ * C_);
  k_transpose<<<(C_ * N1_ + 255) / 256, 256, 0, stream>>>(w_attn, wT, C_, N1_);
  k_transpose<<<(C_ * C_ + 255) / 256, 256, 0, stream>>>(w_proj, wpT, C_, C_);

  gemm_qkv<<<dim3(M_ / 128, N1_ / 128), 256, 0, stream>>>(xbf, wT, b_attn, Qb, Kb, Vtb);
  attn_fwd<<<dim3(T_ / 64, B_ * H_), 128, 0, stream>>>(Qb, Kb, Vtb, Yb);
  gemm_proj<<<dim3(M_ / 128, C_ / 128), 256, 0, stream>>>(Yb, wpT, b_proj, out);
}
